// QNet_10806137717289
// MI455X (gfx1250) — matrix operand order repaired and checked by reading, not compiled, not run on hardware
//
#include <hip/hip_runtime.h>

// ---------------- problem constants ----------------
#define BATCH     65536
#define OBS_DIM   1129
#define N_OTHERS  39
#define N_ATT     40
#define EM        32
#define HID       32
#define N_ACT     21

// padded f16 weight layouts (transposed: WT[n][k], row length = KPAD)
#define KPAD_AL   1152   // 36 * 32
#define NPAD_AL   48
#define KPAD_SELF 64
#define KPAD_32   32

#define WAVES_PER_BLOCK 8
#define BLOCK_THREADS   (WAVES_PER_BLOCK * 32)
#define TILES           (BATCH / 16)                 // 4096
#define GRID_BLOCKS     (TILES / WAVES_PER_BLOCK)    // 512

// LDS layout (dynamic shared memory)
#define SMEM_WAL_HALVES   (NPAD_AL * KPAD_AL)                    // 55296
#define SMEM_SMALL_HALVES (2048 + 1024 + 1024 + 1024 + 1024)     // 6144
#define SMEM_WAL_BYTES    (SMEM_WAL_HALVES * 2)                  // 110592
#define SMEM_SMALL_BYTES  (SMEM_SMALL_HALVES * 2)                // 12288
#define SMEM_SCRATCH_BYTES (WAVES_PER_BLOCK * 16 * 32 * 4)       // 16384
#define SMEM_TOTAL        (SMEM_WAL_BYTES + SMEM_SMALL_BYTES + SMEM_SCRATCH_BYTES)

typedef __attribute__((ext_vector_type(16))) _Float16 v16h;
typedef __attribute__((ext_vector_type(8)))  _Float16 v8h;
typedef __attribute__((ext_vector_type(8)))  float    v8f;
typedef __attribute__((ext_vector_type(4)))  unsigned uint32x4;
typedef __attribute__((ext_vector_type(4)))  int      int32x4;
typedef __attribute__((ext_vector_type(8)))  int      int32x8;

#define AS1 __attribute__((address_space(1)))
#define AS3 __attribute__((address_space(3)))

#if __has_builtin(__builtin_amdgcn_global_load_async_to_lds_b128)
#define HAVE_ASYNC_LDS 1
#endif
#if __has_builtin(__builtin_amdgcn_tensor_load_to_lds)
#define HAVE_TDM 1
#endif

// ---------------- WMMA helper ----------------
static __device__ __forceinline__ v8f wmma_f16(v16h a, v16h b, v8f c) {
  // D = A(16x32 f16) * B(32x16 f16) + C(16x16 f32)
  return __builtin_amdgcn_wmma_f32_16x16x32_f16(false, a, false, b, (short)0, c,
                                                false, false);
}

// stage nhalves f16 elements global -> LDS (async if available)
static __device__ __forceinline__ void stage16(_Float16* ldst,
                                               const _Float16* gsrc,
                                               int nhalves, int t, int nt) {
  for (int i = t; i < (nhalves >> 3); i += nt) {
#if defined(HAVE_ASYNC_LDS)
    __builtin_amdgcn_global_load_async_to_lds_b128(
        (AS1 int32x4*)(unsigned long long)(gsrc + (size_t)i * 8),
        (AS3 int32x4*)(ldst + (size_t)i * 8), 0, 0);
#else
    *(v8h*)(ldst + (size_t)i * 8) = *(const v8h*)(gsrc + (size_t)i * 8);
#endif
  }
}

static __device__ __forceinline__ void wait_async_staging() {
#if defined(HAVE_ASYNC_LDS)
#if __has_builtin(__builtin_amdgcn_s_wait_asynccnt)
  __builtin_amdgcn_s_wait_asynccnt(0);
#else
  asm volatile("s_wait_asynccnt 0x0" ::: "memory");
#endif
#endif
}

// A operand (16x32 f16) from an fp32 row pointer, K window [kbase, kbase+kcount)
static __device__ __forceinline__ v16h load_a_f32(const float* __restrict__ rowp,
                                                  int kbase, int kcount, int hi) {
  v16h a;
  const int b0 = hi * 16;
  const int b1 = hi * 16 + 8;
#pragma unroll
  for (int h = 0; h < 8; ++h) {
    const int r0 = b0 + h;
    const int r1 = b1 + h;
    a[h]     = (r0 < kcount) ? (_Float16)rowp[kbase + r0] : (_Float16)0.0f;
    a[h + 8] = (r1 < kcount) ? (_Float16)rowp[kbase + r1] : (_Float16)0.0f;
  }
  return a;
}

// B operand (32x16 f16) from pre-transposed padded f16 weights WT[n][k]
// (works for both global and LDS pointers; 32B contiguous per lane)
static __device__ __forceinline__ v16h load_b_t(const _Float16* wT, int kpad,
                                                int ntile, int kchunk, int lane) {
  const int n = (lane & 15) + (ntile << 4);
  const int k = (kchunk << 5) + ((lane >> 4) << 4);
  return *(const v16h*)(wT + (size_t)n * kpad + k);
}

// C-pair (16 rows x 32 cols f32) -> LDS scratch [16][32]
static __device__ __forceinline__ void c_pair_to_lds(float (*sc)[32],
                                                     const v8f c0, const v8f c1,
                                                     int lane) {
  const int r0 = (lane >> 4) << 3;
  const int cc = lane & 15;
#pragma unroll
  for (int r = 0; r < 8; ++r) {
    sc[r0 + r][cc]      = c0[r];
    sc[r0 + r][cc + 16] = c1[r];
  }
}

// A operand (16x32 f16) from LDS scratch [16][32] f32
static __device__ __forceinline__ v16h a_from_lds(const float (*sc)[32], int lane) {
  v16h a;
  const int row = lane & 15;
  const int b0  = (lane >> 4) << 4;
#pragma unroll
  for (int h = 0; h < 8; ++h) {
    a[h]     = (_Float16)sc[row][b0 + h];
    a[h + 8] = (_Float16)sc[row][b0 + 8 + h];
  }
  return a;
}

// per-agent encoder accumulation for agent indices a = nbeg+1 .. nend
static __device__ __forceinline__ void accum_others(
    const float* __restrict__ rowp, int hi, int lane, int srcbase,
    int nbeg, int nend, const v8f wt, v16h bo0v, v16h bo1v,
    float bbo0, float bbo1, v8f& ao0, v8f& ao1) {
  for (int n = nbeg; n < nend; ++n) {
    v16h a = load_a_f32(rowp, 37 + 28 * n, 28, hi);
    v8f z = {};
    v8f o0 = wmma_f16(a, bo0v, z);
    v8f o1 = wmma_f16(a, bo1v, z);
    const int src = srcbase | ((n + 1) & 15);
#pragma unroll
    for (int r = 0; r < 8; ++r) {
      const float wv = __shfl(wt[r], src, 32);
      ao0[r] += wv * fmaxf(o0[r] + bbo0, 0.0f);
      ao1[r] += wv * fmaxf(o1[r] + bbo1, 0.0f);
    }
  }
}

// ---------------- weight prep: fp32 [K][N] -> f16 transposed padded [Npad][Kpad] ----
__global__ void qnet_prep_weights(const float* __restrict__ src,
                                  _Float16* __restrict__ dst,
                                  int K, int N, int Kpad, int Npad) {
  const int i = blockIdx.x * blockDim.x + threadIdx.x;
  const int total = Kpad * Npad;
  if (i >= total) return;
  const int n = i / Kpad;
  const int k = i - n * Kpad;
  const float v = (n < N && k < K) ? src[k * N + n] : 0.0f;
  dst[(size_t)n * Kpad + k] = (_Float16)v;
}

// ---------------- main fused kernel ----------------
__global__ __launch_bounds__(BLOCK_THREADS) void qnet_main(
    const float* __restrict__ obs,
    const float* __restrict__ b_al,    const float* __restrict__ b_self,
    const float* __restrict__ b_other, const float* __restrict__ b_attout,
    const float* __restrict__ b_emb,   const float* __restrict__ b_out,
    const _Float16* __restrict__ walT,    const _Float16* __restrict__ wselfT,
    const _Float16* __restrict__ wotherT, const _Float16* __restrict__ wattoutT,
    const _Float16* __restrict__ wembT,   const _Float16* __restrict__ woutT,
    float* __restrict__ out) {
  extern __shared__ char qnet_smem[];
  _Float16* walL     = (_Float16*)qnet_smem;
  _Float16* wselfL   = (_Float16*)(qnet_smem + SMEM_WAL_BYTES);
  _Float16* wotherL  = wselfL + 2048;
  _Float16* wattoutL = wotherL + 1024;
  _Float16* wembL    = wattoutL + 1024;
  _Float16* woutL    = wembL + 1024;

  const int tid  = threadIdx.x;
  const int lane = tid & 31;
  const int wave = tid >> 5;
  const int tile = blockIdx.x * WAVES_PER_BLOCK + wave;
  const int m0   = tile * 16;

  float (*scratch)[32] =
      (float (*)[32])(qnet_smem + SMEM_WAL_BYTES + SMEM_SMALL_BYTES) + wave * 16;

  // ---- stage weights on-chip ----
#if defined(HAVE_TDM)
  if (wave == 0) {
    // 2-D TDM descriptor: 48 x 1152 f16 tensor == tile, contiguous
    const unsigned long long ga = (unsigned long long)walT;
    const unsigned lds_off = (unsigned)(unsigned long long)(AS3 char*)qnet_smem;
    uint32x4 g0 = {};
    g0[0] = 1u;                                          // count=1 user D#
    g0[1] = lds_off;                                     // lds_addr
    g0[2] = (unsigned)(ga & 0xFFFFFFFFull);              // global_addr[31:0]
    g0[3] = (unsigned)((ga >> 32) & 0x1FFFFFFull)        // global_addr[56:32]
            | 0x80000000u;                               // type=2 (image)
    int32x8 g1 = {};
    g1[0] = 0x00010000;                                  // data_size=1 (2 bytes)
    g1[1] = (int)((unsigned)KPAD_AL << 16);              // tensor_dim0 lo16
    g1[2] = (int)((unsigned)NPAD_AL << 16);              // tensor_dim1 lo16
    g1[3] = (int)((unsigned)KPAD_AL << 16);              // tile_dim0
    g1[4] = NPAD_AL;                                     // tile_dim1 (tile_dim2=0)
    g1[5] = KPAD_AL;                                     // tensor_dim0_stride lo32
    int32x4 gz = {};
#if __clang_major__ >= 23
    int32x8 gz8 = {};
    __builtin_amdgcn_tensor_load_to_lds(g0, g1, gz, gz, gz8, 0);
#else
    __builtin_amdgcn_tensor_load_to_lds(g0, g1, gz, gz, 0);
#endif
  }
#else
  stage16(walL, walT, SMEM_WAL_HALVES, tid, BLOCK_THREADS);
#endif
  stage16(wselfL,   wselfT,   2048, tid, BLOCK_THREADS);
  stage16(wotherL,  wotherT,  1024, tid, BLOCK_THREADS);
  stage16(wattoutL, wattoutT, 1024, tid, BLOCK_THREADS);
  stage16(wembL,    wembT,    1024, tid, BLOCK_THREADS);
  stage16(woutL,    woutT,    1024, tid, BLOCK_THREADS);
  wait_async_staging();
#if defined(HAVE_TDM)
  __builtin_amdgcn_s_wait_tensorcnt(0);
#endif
  __syncthreads();

  const int col     = lane & 15;
  const int hi      = lane >> 4;
  const int rowsel  = hi * 8;
  const int srcbase = lane & 16;

  const float* rowp = obs + (size_t)(m0 + col) * OBS_DIM;

  // ---- phase 1: attention logits (16x1129 @ 1129x40) + self encoder ----
  v8f l0 = {}, l1 = {}, l2 = {};
  v8f s0 = {}, s1 = {};
  for (int c = 0; c < 36; ++c) {
    const int kb = c * 32;
    const int kc = (OBS_DIM - kb) < 32 ? (OBS_DIM - kb) : 32;
    __builtin_prefetch(rowp + kb + 128, 0, 1);
    v16h a = load_a_f32(rowp, kb, kc, hi);
    l0 = wmma_f16(a, load_b_t(walL, KPAD_AL, 0, c, lane), l0);
    l1 = wmma_f16(a, load_b_t(walL, KPAD_AL, 1, c, lane), l1);
    l2 = wmma_f16(a, load_b_t(walL, KPAD_AL, 2, c, lane), l2);
    if (c < 2) {  // self encoder: K = 0..36 (rows 37..63 of wselfL are zero)
      s0 = wmma_f16(a, load_b_t(wselfL, KPAD_SELF, 0, c, lane), s0);
      s1 = wmma_f16(a, load_b_t(wselfL, KPAD_SELF, 1, c, lane), s1);
    }
  }

  // ---- phase 2: bias + row softmax over 40 attention columns ----
  const bool  v2   = (col < 8);
  const float bal0 = b_al[col];
  const float bal1 = b_al[16 + col];
  const float bal2 = v2 ? b_al[32 + col] : 0.0f;
  const float NEG = -3.0e38f;

  v8f w0, w1, w2;
#pragma unroll
  for (int r = 0; r < 8; ++r) {
    const float x0 = l0[r] + bal0;
    const float x1 = l1[r] + bal1;
    const float x2 = v2 ? (l2[r] + bal2) : NEG;
    float m = fmaxf(fmaxf(x0, x1), x2);
#pragma unroll
    for (int mask = 8; mask >= 1; mask >>= 1) m = fmaxf(m, __shfl_xor(m, mask, 32));
    const float e0 = __expf(x0 - m);
    const float e1 = __expf(x1 - m);
    const float e2 = v2 ? __expf(x2 - m) : 0.0f;
    float s = e0 + e1 + e2;
#pragma unroll
    for (int mask = 8; mask >= 1; mask >>= 1) s += __shfl_xor(s, mask, 32);
    const float inv = __frcp_rn(s);
    w0[r] = e0 * inv; w1[r] = e1 * inv; w2[r] = e2 * inv;
  }

  // ---- phase 3: att_out = sum_a att_w[:,a] * enc[:,a,:] ----
  const float bs0 = b_self[col];
  const float bs1 = b_self[16 + col];
  v8f ao0, ao1;
#pragma unroll
  for (int r = 0; r < 8; ++r) {   // a = 0 : self encoding (no relu)
    const float wv = __shfl(w0[r], srcbase, 32);
    ao0[r] = wv * (s0[r] + bs0);
    ao1[r] = wv * (s1[r] + bs1);
  }

  v16h bo0v = load_b_t(wotherL, KPAD_32, 0, 0, lane);
  v16h bo1v = load_b_t(wotherL, KPAD_32, 1, 0, lane);
  const float bbo0 = b_other[col];
  const float bbo1 = b_other[16 + col];
  accum_others(rowp, hi, lane, srcbase,  0, 15, w0, bo0v, bo1v, bbo0, bbo1, ao0, ao1);
  accum_others(rowp, hi, lane, srcbase, 15, 31, w1, bo0v, bo1v, bbo0, bbo1, ao0, ao1);
  accum_others(rowp, hi, lane, srcbase, 31, 39, w2, bo0v, bo1v, bbo0, bbo1, ao0, ao1);

  // ---- phase 4: dense head via LDS transposes ----
  v8f z = {};

  c_pair_to_lds(scratch, ao0, ao1, lane);
  __syncthreads();
  v16h ha = a_from_lds(scratch, lane);
  __syncthreads();
  v8f h0 = wmma_f16(ha, load_b_t(wattoutL, KPAD_32, 0, 0, lane), z);
  v8f h1 = wmma_f16(ha, load_b_t(wattoutL, KPAD_32, 1, 0, lane), z);
  {
    const float b0v = b_attout[col], b1v = b_attout[16 + col];
#pragma unroll
    for (int r = 0; r < 8; ++r) {
      h0[r] = fmaxf(h0[r] + b0v, 0.0f);
      h1[r] = fmaxf(h1[r] + b1v, 0.0f);
    }
  }

  c_pair_to_lds(scratch, h0, h1, lane);
  __syncthreads();
  ha = a_from_lds(scratch, lane);
  __syncthreads();
  v8f g0 = wmma_f16(ha, load_b_t(wembL, KPAD_32, 0, 0, lane), z);
  v8f g1 = wmma_f16(ha, load_b_t(wembL, KPAD_32, 1, 0, lane), z);
  {
    const float b0v = b_emb[col], b1v = b_emb[16 + col];
#pragma unroll
    for (int r = 0; r < 8; ++r) {
      g0[r] = fmaxf(g0[r] + b0v, 0.0f);
      g1[r] = fmaxf(g1[r] + b1v, 0.0f);
    }
  }

  c_pair_to_lds(scratch, g0, g1, lane);
  __syncthreads();
  ha = a_from_lds(scratch, lane);
  __syncthreads();
  v8f q0 = wmma_f16(ha, load_b_t(woutL, KPAD_32, 0, 0, lane), z);
  v8f q1 = wmma_f16(ha, load_b_t(woutL, KPAD_32, 1, 0, lane), z);

  const float bq0 = b_out[col];
  const bool  vq1 = (col < (N_ACT - 16));
  const float bq1 = vq1 ? b_out[16 + col] : 0.0f;
#pragma unroll
  for (int r = 0; r < 8; ++r) {
    const int row = m0 + rowsel + r;
    __builtin_nontemporal_store(q0[r] + bq0, &out[(size_t)row * N_ACT + col]);
    if (vq1)
      __builtin_nontemporal_store(q1[r] + bq1, &out[(size_t)row * N_ACT + 16 + col]);
  }
}

// ---------------- host-side launch ----------------
extern "C" void kernel_launch(void* const* d_in, const int* in_sizes, int n_in,
                              void* d_out, int out_size, void* d_ws, size_t ws_size,
                              hipStream_t stream) {
  (void)in_sizes; (void)n_in; (void)out_size; (void)ws_size;

  const float* obs      = (const float*)d_in[0];
  const float* W_al     = (const float*)d_in[1];
  const float* b_al     = (const float*)d_in[2];
  const float* W_self   = (const float*)d_in[3];
  const float* b_self   = (const float*)d_in[4];
  const float* W_other  = (const float*)d_in[5];
  const float* b_other  = (const float*)d_in[6];
  const float* W_attout = (const float*)d_in[7];
  const float* b_attout = (const float*)d_in[8];
  const float* W_emb    = (const float*)d_in[9];
  const float* b_emb    = (const float*)d_in[10];
  const float* W_out    = (const float*)d_in[11];
  const float* b_out    = (const float*)d_in[12];
  float* out = (float*)d_out;

  _Float16* w16 = (_Float16*)d_ws;
  _Float16* walT     = w16;                                   // 48 * 1152
  _Float16* wselfT   = walT    + (size_t)NPAD_AL * KPAD_AL;   // 32 * 64
  _Float16* wotherT  = wselfT  + (size_t)32 * KPAD_SELF;      // 32 * 32
  _Float16* wattoutT = wotherT + (size_t)32 * KPAD_32;
  _Float16* wembT    = wattoutT+ (size_t)32 * KPAD_32;
  _Float16* woutT    = wembT   + (size_t)32 * KPAD_32;

  const int al_total = NPAD_AL * KPAD_AL;
  qnet_prep_weights<<<(al_total + 255) / 256, 256, 0, stream>>>(
      W_al, walT, OBS_DIM, N_ATT, KPAD_AL, NPAD_AL);
  qnet_prep_weights<<<(32 * KPAD_SELF + 255) / 256, 256, 0, stream>>>(
      W_self, wselfT, 37, EM, KPAD_SELF, 32);
  qnet_prep_weights<<<(32 * KPAD_32 + 255) / 256, 256, 0, stream>>>(
      W_other, wotherT, 28, EM, KPAD_32, 32);
  qnet_prep_weights<<<(32 * KPAD_32 + 255) / 256, 256, 0, stream>>>(
      W_attout, wattoutT, EM, HID, KPAD_32, 32);
  qnet_prep_weights<<<(32 * KPAD_32 + 255) / 256, 256, 0, stream>>>(
      W_emb, wembT, HID, HID, KPAD_32, 32);
  qnet_prep_weights<<<(32 * KPAD_32 + 255) / 256, 256, 0, stream>>>(
      W_out, woutT, HID, N_ACT, KPAD_32, 32);

  (void)hipFuncSetAttribute((const void*)qnet_main,
                            hipFuncAttributeMaxDynamicSharedMemorySize,
                            SMEM_TOTAL);
  qnet_main<<<GRID_BLOCKS, BLOCK_THREADS, SMEM_TOTAL, stream>>>(
      obs, b_al, b_self, b_other, b_attout, b_emb, b_out,
      walT, wselfT, wotherT, wattoutT, wembT, woutT, out);
}
